// VQVAE_all_35828617183682
// MI455X (gfx1250) — matrix operand order repaired and checked by reading, not compiled, not run on hardware
//
#include <hip/hip_runtime.h>
#include <hip/hip_bf16.h>
#include <math.h>

// Problem constants (from reference)
#define B_DIM   32
#define C_DIM   256
#define D_DIM   256
#define HW      1024           // 32*32
#define NPIX    32768          // B*HW
#define K_CODES 4096
#define OUT_Q_ELEMS 8388608    // B*D*HW
#define STAGE_CODES 32         // codes staged into LDS per double-buffer step

typedef __attribute__((ext_vector_type(16))) _Float16 v16h;
typedef __attribute__((ext_vector_type(8)))  _Float16 v8h;
typedef __attribute__((ext_vector_type(8)))  float    v8f;

// CDNA5 async copy: global -> LDS, 16B per lane, tracked by ASYNCcnt.
// Operands per CDNA5 ISA: VDST = LDS byte address VGPR,
// VADDR = 64-bit global address VGPR pair ("GV" mode).
__device__ __forceinline__ void async_copy_b128(uint32_t lds_off, const void* gaddr)
{
    asm volatile("global_load_async_to_lds_b128 %0, %1, off"
                 :: "v"(lds_off), "v"(gaddr) : "memory");
}

// ---------------------------------------------------------------------------
// K1: 1x1 conv  z[n,d] = sum_c x[b,c,hw]*w[d,c] + bias[d], store f16 [N,D]
// block = 256 threads, handles 32 pixels x all 256 d. Memory-bound; VALU f32.
// ---------------------------------------------------------------------------
__global__ __launch_bounds__(256) void conv1x1_kernel(
    const float* __restrict__ x, const float* __restrict__ w,
    const float* __restrict__ bias, _Float16* __restrict__ zh)
{
    __shared__ float xs[32 * 65];          // [pixel][c-chunk], padded stride 65
    const int tid = threadIdx.x;
    const int n0  = blockIdx.x * 32;
    const int b   = n0 >> 10;              // n0 / HW
    const int hw0 = n0 & 1023;

    float acc[32];
#pragma unroll
    for (int p = 0; p < 32; ++p) acc[p] = 0.f;

    const int d = tid;                     // each thread owns one output dim
    for (int c0 = 0; c0 < C_DIM; c0 += 64) {
        __syncthreads();
        {
            const int cbase = (tid >> 5) << 3;   // 0,8,...,56
            const int p     = tid & 31;
#pragma unroll
            for (int j = 0; j < 8; ++j) {
                const int cc = cbase + j;
                xs[p * 65 + cc] = x[((size_t)(b * C_DIM + c0 + cc) << 10) + hw0 + p];
            }
        }
        __syncthreads();
#pragma unroll 8
        for (int cc = 0; cc < 64; ++cc) {
            const float wv = w[d * C_DIM + c0 + cc];
#pragma unroll
            for (int p = 0; p < 32; ++p)
                acc[p] = fmaf(wv, xs[p * 65 + cc], acc[p]);
        }
    }
    const float bv = bias[d];
#pragma unroll
    for (int p = 0; p < 32; ++p)
        zh[(size_t)(n0 + p) * D_DIM + d] = (_Float16)(acc[p] + bv);
}

// ---------------------------------------------------------------------------
// K2: embedding prep — f32 -> f16 copy + e2[k] = ||emb[k]||^2 (wave per code)
// ---------------------------------------------------------------------------
__global__ __launch_bounds__(256) void emb_prep_kernel(
    const float* __restrict__ emb, _Float16* __restrict__ embh,
    float* __restrict__ e2)
{
    const int wave = threadIdx.x >> 5;
    const int lane = threadIdx.x & 31;
    const int k    = blockIdx.x * 8 + wave;
    const float*   row  = emb  + (size_t)k * D_DIM + lane * 8;
    _Float16*      orow = embh + (size_t)k * D_DIM + lane * 8;
    float s = 0.f;
#pragma unroll
    for (int j = 0; j < 8; ++j) {
        const float v = row[j];
        s = fmaf(v, v, s);
        orow[j] = (_Float16)v;
    }
#pragma unroll
    for (int m = 16; m >= 1; m >>= 1) s += __shfl_xor(s, m, 32);
    if (lane == 0) e2[k] = s;
}

// ---------------------------------------------------------------------------
// K3: WMMA distance + argmin with async-LDS codebook staging.
// Block = 8 waves, each wave owns a 16-pixel M-tile (A in registers).
// Codebook is staged 32 codes (16KB) at a time into double-buffered LDS via
// global_load_async_to_lds_b128; all 8 waves share each stage (8x less L2
// traffic). dist_rel = ||e||^2 - 2*(z.e)  (||z||^2 constant per row: dropped)
// ---------------------------------------------------------------------------
__global__ __launch_bounds__(256) void dist_argmin_kernel(
    const _Float16* __restrict__ zh, const _Float16* __restrict__ embh,
    const float* __restrict__ e2, int* __restrict__ idx, int* __restrict__ hist)
{
    __shared__ _Float16 embS[2][STAGE_CODES * D_DIM];   // 2 x 16 KB

    const int tid     = threadIdx.x;
    const int lane    = tid & 31;
    const int wave    = tid >> 5;
    const int pixBase = blockIdx.x * 128 + wave * 16;
    const int row     = lane & 15;
    const int half    = lane >> 4;

    // A fragments: 16x32 f16 each, 8 fragments cover D=256 (ISA 7.12.2):
    // lanes 0-15 rows M=0..15 hold K {0..7,16..23}; lanes 16-31 hold
    // K {8..15,24..31} of the same rows.
    v16h a[8];
    {
        const _Float16* pz = zh + (size_t)(pixBase + row) * D_DIM + half * 8;
#pragma unroll
        for (int f = 0; f < 8; ++f) {
            const v8h lo = *(const v8h*)(pz + f * 32);
            const v8h hi = *(const v8h*)(pz + f * 32 + 16);
#pragma unroll
            for (int i = 0; i < 8; ++i) { a[f][i] = lo[i]; a[f][i + 8] = hi[i]; }
        }
    }

    // Async stage issue: 16 KB per stage, 256 threads x 4 x 16B, coalesced.
    const char* gbase = (const char*)embh;
    const uint32_t lds0 = (uint32_t)(size_t)(&embS[0][0]);
    const uint32_t lds1 = (uint32_t)(size_t)(&embS[1][0]);
#define ISSUE_STAGE(stage, ldsbase)                                          \
    {                                                                        \
        const char* g = gbase + (size_t)(stage) * (STAGE_CODES * D_DIM * 2); \
        _Pragma("unroll")                                                    \
        for (int i = 0; i < 4; ++i) {                                        \
            const uint32_t off = (uint32_t)tid * 16u + (uint32_t)i * 4096u;  \
            async_copy_b128((ldsbase) + off, g + off);                       \
        }                                                                    \
    }

    ISSUE_STAGE(0, lds0);

    float best[8];
    int   bidx[8];
#pragma unroll
    for (int r = 0; r < 8; ++r) { best[r] = 3.4e38f; bidx[r] = 0; }

    const int NSTAGES = K_CODES / STAGE_CODES;   // 128
    for (int s = 0; s < NSTAGES; ++s) {
        const int buf = s & 1;
        if (s + 1 < NSTAGES) {
            if (buf) { ISSUE_STAGE(s + 1, lds0); }
            else     { ISSUE_STAGE(s + 1, lds1); }
            asm volatile("s_wait_asynccnt 4" ::: "memory");  // stage s landed
        } else {
            asm volatile("s_wait_asynccnt 0" ::: "memory");
        }
        __syncthreads();

#pragma unroll
        for (int sub = 0; sub < 2; ++sub) {
            const int codeLocal = sub * 16 + row;          // this lane's column
            const int code      = s * STAGE_CODES + codeLocal;
            // B fragments from LDS: two 16B chunks per lane per fragment (K {0..7,16..23} / {8..15,24..31}).
            const _Float16* pe = &embS[buf][codeLocal * D_DIM + half * 8];
            v16h bf[8];
#pragma unroll
            for (int f = 0; f < 8; ++f) {
                const v8h lo = *(const v8h*)(pe + f * 32);
                const v8h hi = *(const v8h*)(pe + f * 32 + 16);
#pragma unroll
                for (int i = 0; i < 8; ++i) { bf[f][i] = lo[i]; bf[f][i + 8] = hi[i]; }
            }

            v8f c = {0.f, 0.f, 0.f, 0.f, 0.f, 0.f, 0.f, 0.f};
#pragma unroll
            for (int f = 0; f < 8; ++f)
                c = __builtin_amdgcn_wmma_f32_16x16x32_f16(
                        false, a[f], false, bf[f], (short)0, c, false, false);

            const float ev = e2[code];
#pragma unroll
            for (int r = 0; r < 8; ++r) {
                const float dv = ev - 2.0f * c[r];
                if (dv < best[r]) { best[r] = dv; bidx[r] = code; }
            }
        }
        __syncthreads();   // protect buf from next iteration's async overwrite
    }
#undef ISSUE_STAGE

    // Cross-lane argmin within each 16-lane half (xor 8,4,2,1 stays in-half).
#pragma unroll
    for (int r = 0; r < 8; ++r) {
#pragma unroll
        for (int m = 8; m >= 1; m >>= 1) {
            const float ob = __shfl_xor(best[r], m, 32);
            const int   oi = __shfl_xor(bidx[r], m, 32);
            if (ob < best[r] || (ob == best[r] && oi < bidx[r])) {
                best[r] = ob; bidx[r] = oi;
            }
        }
    }
    // C layout: VGPR r -> M=r (lanes 0-15) / M=r+8 (lanes 16-31)
    if (row == 0) {
#pragma unroll
        for (int r = 0; r < 8; ++r) {
            const int m = half * 8 + r;
            idx[pixBase + m] = bidx[r];
            atomicAdd(&hist[bidx[r]], 1);
        }
    }
}

// ---------------------------------------------------------------------------
// K4: gather quantized = emb[idx], transpose to [B,D,H,W] (coalesced writes),
// fused SSE accumulation for the loss.
// ---------------------------------------------------------------------------
__global__ __launch_bounds__(256) void gather_loss_kernel(
    const float* __restrict__ emb, const _Float16* __restrict__ zh,
    const int* __restrict__ idx, float* __restrict__ out, float* __restrict__ sse)
{
    __shared__ float red[256];
    const int tid    = threadIdx.x;
    const int hwl    = tid & 63;
    const int dq     = tid >> 6;
    const int n0     = blockIdx.x * 64;
    const int b      = n0 >> 10;
    const int hwbase = n0 & 1023;
    const int n      = n0 + hwl;

    const int code = idx[n];
    const float*    erow = emb + (size_t)code * D_DIM;
    const _Float16* zrow = zh  + (size_t)n * D_DIM;
    const size_t obase = (size_t)b * (D_DIM * HW) + hwbase + hwl;

    float s = 0.f;
    for (int d = dq; d < D_DIM; d += 4) {
        const float q  = erow[d];
        const float zv = (float)zrow[d];
        const float df = q - zv;
        s = fmaf(df, df, s);
        out[obase + (size_t)d * HW] = q;   // straight-through fwd value == q
    }
    red[tid] = s;
    __syncthreads();
    for (int st = 128; st > 0; st >>= 1) {
        if (tid < st) red[tid] += red[tid + st];
        __syncthreads();
    }
    if (tid == 0) atomicAdd(sse, red[0]);
}

// ---------------------------------------------------------------------------
// K5: finalize loss (1.25 * MSE) and perplexity from histogram.
// ---------------------------------------------------------------------------
__global__ __launch_bounds__(256) void finalize_kernel(
    const int* __restrict__ hist, const float* __restrict__ sse,
    float* __restrict__ out)
{
    __shared__ float red[256];
    const int tid = threadIdx.x;
    float ent = 0.f;
    for (int k = tid; k < K_CODES; k += 256) {
        const float p = (float)hist[k] * (1.0f / (float)NPIX);
        ent -= p * logf(p + 1e-10f);
    }
    red[tid] = ent;
    __syncthreads();
    for (int st = 128; st > 0; st >>= 1) {
        if (tid < st) red[tid] += red[tid + st];
        __syncthreads();
    }
    if (tid == 0) {
        out[OUT_Q_ELEMS]     = 1.25f * sse[0] / (float)OUT_Q_ELEMS; // q + 0.25*e
        out[OUT_Q_ELEMS + 1] = expf(red[0]);
    }
}

// ---------------------------------------------------------------------------
extern "C" void kernel_launch(void* const* d_in, const int* in_sizes, int n_in,
                              void* d_out, int out_size, void* d_ws, size_t ws_size,
                              hipStream_t stream)
{
    const float* x      = (const float*)d_in[0];
    const float* conv_w = (const float*)d_in[1];
    const float* conv_b = (const float*)d_in[2];
    const float* emb    = (const float*)d_in[3];
    float* out = (float*)d_out;

    // Workspace layout (19,038,224 bytes total)
    char* w = (char*)d_ws;
    _Float16* zh   = (_Float16*)(w);                         // 16,777,216 B
    _Float16* embh = (_Float16*)(w + 16777216);              //  2,097,152 B
    float*    e2   = (float*)   (w + 18874368);              //     16,384 B
    int*      idxb = (int*)     (w + 18890752);              //    131,072 B
    int*      hist = (int*)     (w + 19021824);              //     16,384 B
    float*    sse  = (float*)   (w + 19038208);              //         16 B

    // zero hist + sse (contiguous) — capture-safe
    hipMemsetAsync(hist, 0, K_CODES * sizeof(int) + 16, stream);

    conv1x1_kernel    <<<NPIX / 32,  256, 0, stream>>>(x, conv_w, conv_b, zh);
    emb_prep_kernel   <<<K_CODES / 8, 256, 0, stream>>>(emb, embh, e2);
    dist_argmin_kernel<<<NPIX / 128, 256, 0, stream>>>(zh, embh, e2, idxb, hist);
    gather_loss_kernel<<<NPIX / 64,  256, 0, stream>>>(emb, zh, idxb, out, sse);
    finalize_kernel   <<<1,          256, 0, stream>>>(hist, sse, out);
}
